// MultiHeadAttention_23673859736398
// MI455X (gfx1250) — matrix operand order repaired and checked by reading, not compiled, not run on hardware
//
#include <hip/hip_runtime.h>

#define Bn 4
#define Sn 2048
#define En 512
#define Hn 8
#define Dn 64

typedef _Float16 half_t;
typedef __attribute__((ext_vector_type(16))) _Float16 v16h;
typedef __attribute__((ext_vector_type(8)))  _Float16 v8h;
typedef __attribute__((ext_vector_type(8)))  float    v8f;

static __device__ inline v8f v8f_zero() {
  v8f z;
#pragma unroll
  for (int i = 0; i < 8; ++i) z[i] = 0.f;
  return z;
}

static __device__ inline v8f wmma_f16(v16h a, v16h b, v8f c) {
  // D = A(16xK f16) * B(Kx16 f16) + C(16x16 f32), K=32
  return __builtin_amdgcn_wmma_f32_16x16x32_f16(false, a, false, b, (short)0, c,
                                                false, false);
}

static __device__ inline v16h cat8(v8h lo, v8h hi) {
  v16h r;
#pragma unroll
  for (int i = 0; i < 8; ++i) { r[i] = lo[i]; r[8 + i] = hi[i]; }
  return r;
}

// A-operand (16x32 f16). row = this lane's matrix row (M = lane&15).
// Lanes 0-15 hold K = kk+{0..7, 16..23}; lanes 16-31 hold K = kk+{8..15, 24..31}.
static __device__ inline v16h load_a16(const half_t* row, int kk, int hi) {
  const half_t* p = row + kk + hi * 8;
  v8h lo = *(const v8h*)(p);
  v8h h2 = *(const v8h*)(p + 16);
  return cat8(lo, h2);
}

// Same A layout, converting from f32 source in-lane.
static __device__ inline v16h load_a32(const float* __restrict__ row, int kk, int hi) {
  const float* p = row + kk + hi * 8;
  v16h r;
#pragma unroll
  for (int i = 0; i < 8; ++i) {
    r[i]     = (half_t)p[i];
    r[8 + i] = (half_t)p[16 + i];
  }
  return r;
}

// B-operand (32x16 f16). colp = column n's K-contiguous data (n = lane&15).
// Lanes 0-15 hold K = kk+{0..7, 16..23}; lanes 16-31 hold K = kk+{8..15, 24..31}.
static __device__ inline v16h load_b16(const half_t* colp, int kk, int hi) {
  const half_t* p = colp + kk + hi * 8;
  return cat8(*(const v8h*)(p), *(const v8h*)(p + 16));
}

// ---- CDNA5 async global->LDS (per-lane 16B), ASYNCcnt-tracked -------------
static __device__ inline unsigned lds_off(const void* p) {
  // Flat LDS addresses carry the aperture in the high 32 bits; low 32 bits are
  // the wave-relative LDS byte offset (ISA 10.2 aperture mapping).
  return (unsigned)(unsigned long long)p;
}
static __device__ inline void async_g2l_b128(unsigned ldsoff, const void* g) {
  asm volatile("global_load_async_to_lds_b128 %0, %1, off"
               :: "v"(ldsoff), "v"(g) : "memory");
}
static __device__ inline void wait_async_le2() {
  asm volatile("s_wait_asynccnt 0x2" ::: "memory");
}
static __device__ inline void wait_async_le1() {
  asm volatile("s_wait_asynccnt 0x1" ::: "memory");
}
static __device__ inline void wait_async_0() {
  asm volatile("s_wait_asynccnt 0x0" ::: "memory");
}
static __device__ inline void wait_ds0() {
  asm volatile("s_wait_dscnt 0x0" ::: "memory");
}

// ---------------------------------------------------------------- weights f32->f16
__global__ void __launch_bounds__(256)
mha_prep_kernel(const float* __restrict__ Wq, const float* __restrict__ Wk,
                const float* __restrict__ Wv, const float* __restrict__ Wfc,
                half_t* __restrict__ wq16, half_t* __restrict__ wk16,
                half_t* __restrict__ wv16, half_t* __restrict__ wfc16) {
  int i = blockIdx.x * blockDim.x + threadIdx.x;
  if (i < Dn * Dn) {
    wq16[i] = (half_t)Wq[i];
    wk16[i] = (half_t)Wk[i];
    wv16[i] = (half_t)Wv[i];
  }
  if (i < En * En) wfc16[i] = (half_t)Wfc[i];
}

// ---------------------------------------------------------------- per-head projections
// Qh, Kh: [B,H,S,D] f16.  Vt: [B,H,D,S] f16 (transposed for the PV B-operand).
// Q/K tiles staged in LDS so the global stores are coalesced b128s.
__global__ void __launch_bounds__(256)
mha_proj_kernel(const float* __restrict__ qin, const float* __restrict__ kin,
                const float* __restrict__ vin,
                const half_t* __restrict__ wq, const half_t* __restrict__ wk,
                const half_t* __restrict__ wv,
                half_t* __restrict__ Qh, half_t* __restrict__ Kh,
                half_t* __restrict__ Vt) {
  __shared__ half_t stq[8][16 * Dn];   // 16KB
  __shared__ half_t stk[8][16 * Dn];   // 16KB

  const int lane = threadIdx.x & 31;
  const int wave = threadIdx.x >> 5;
  const int tile = blockIdx.x * 8 + wave;        // Bn*Hn*(Sn/16) tiles
  const int st   = tile % (Sn / 16);
  const int bh   = tile / (Sn / 16);
  const int s0   = st * 16;
  const int hi   = lane >> 4, ln = lane & 15;

  const int b = bh / Hn, h = bh % Hn;
  const size_t inrow = ((size_t)b * Sn + s0 + ln) * En + (size_t)h * Dn;
  const float* qrow = qin + inrow;
  const float* krow = kin + inrow;
  const float* vrow = vin + inrow;

  // A operands: 16 tokens x 64 dims, two K=32 steps, reused across the 4 N-tiles.
  v16h qa0 = load_a32(qrow, 0, hi), qa1 = load_a32(qrow, 32, hi);
  v16h ka0 = load_a32(krow, 0, hi), ka1 = load_a32(krow, 32, hi);
  v16h va0 = load_a32(vrow, 0, hi), va1 = load_a32(vrow, 32, hi);

#pragma unroll
  for (int nt = 0; nt < 4; ++nt) {
    const int e = nt * 16 + ln;                  // output feature (B-column)
    v8f aq = v8f_zero(), ak = v8f_zero(), av = v8f_zero();
    aq = wmma_f16(qa0, load_b16(wq + (size_t)e * Dn, 0, hi), aq);
    aq = wmma_f16(qa1, load_b16(wq + (size_t)e * Dn, 32, hi), aq);
    ak = wmma_f16(ka0, load_b16(wk + (size_t)e * Dn, 0, hi), ak);
    ak = wmma_f16(ka1, load_b16(wk + (size_t)e * Dn, 32, hi), ak);
    av = wmma_f16(va0, load_b16(wv + (size_t)e * Dn, 0, hi), av);
    av = wmma_f16(va1, load_b16(wv + (size_t)e * Dn, 32, hi), av);

    // C layout: lane holds column e; VGPR r holds token row r + hi*8.
#pragma unroll
    for (int r = 0; r < 8; ++r) {
      stq[wave][(r + hi * 8) * Dn + e] = (half_t)aq[r];
      stk[wave][(r + hi * 8) * Dn + e] = (half_t)ak[r];
    }
    // V transposed: row d = e, contiguous in s -> one packed 16B store.
    v8h pv;
#pragma unroll
    for (int r = 0; r < 8; ++r) pv[r] = (half_t)av[r];
    *(v8h*)(Vt + ((size_t)bh * Dn + e) * Sn + s0 + hi * 8) = pv;
  }

  wait_ds0();  // wave-local: staging writes before readback (LDS is in-order per wave)

  // Coalesced copy-out: 16 rows x 128B are contiguous in Qh/Kh -> 4 x b128 per lane.
  half_t* qdst = Qh + ((size_t)bh * Sn + s0) * Dn;
  half_t* kdst = Kh + ((size_t)bh * Sn + s0) * Dn;
#pragma unroll
  for (int k2 = 0; k2 < 4; ++k2) {
    const int j = (lane + 32 * k2) * 8;          // half offset, 16B granules
    *(v8h*)(qdst + j) = *(const v8h*)&stq[wave][j];
    *(v8h*)(kdst + j) = *(const v8h*)&stk[wave][j];
  }
}

// ---------------------------------------------------------------- flash attention
// Block = 8 waves = 128 consecutive queries of one (b,h). K/V chunks (32 keys)
// are async-DMAed into double-buffered LDS (2 async b128 per thread per chunk),
// overlapped with WMMA on the previous chunk. Scores computed transposed so the
// softmaxed P lands directly in the PV A-operand layout (no cross-lane moves).
__global__ void __launch_bounds__(256)
mha_attn_kernel(const half_t* __restrict__ Qh, const half_t* __restrict__ Kh,
                const half_t* __restrict__ Vt, half_t* __restrict__ Ah) {
  __shared__ half_t kbuf[2][32 * Dn];   // [buf][key][d]   4KB each
  __shared__ half_t vbuf[2][Dn * 32];   // [buf][d][key]   4KB each
  __shared__ half_t obuf[8][16 * Dn];   // output staging  16KB

  const int tid  = threadIdx.x;
  const int lane = tid & 31;
  const int wave = tid >> 5;
  const int qblk = blockIdx.x % (Sn / 128);
  const int bh   = blockIdx.x / (Sn / 128);
  const int b    = bh / Hn, h = bh % Hn;
  const int q0   = qblk * 128 + wave * 16;
  const int hi   = lane >> 4, ln = lane & 15;
  const int nch  = (qblk * 128 + 128) / 32;      // causal chunk count (block-uniform)

  const half_t* kg = Kh + (size_t)bh * Sn * Dn;  // chunk rows are 4KB contiguous
  const half_t* vg = Vt + (size_t)bh * Dn * Sn;  // 64 rows x 64B per chunk
  const int vrow = tid >> 2, vseg = tid & 3;     // V copy: 4 lanes per d-row

  // B-operand for St: B[d][q] = Qhat[q][d]; lane n = query q0+ln (L2-resident).
  const half_t* qrow = Qh + ((size_t)bh * Sn + q0 + ln) * Dn;
  const v16h qb0 = load_b16(qrow, 0, hi);
  const v16h qb1 = load_b16(qrow, 32, hi);

  v8f acc0 = v8f_zero(), acc1 = v8f_zero(), acc2 = v8f_zero(), acc3 = v8f_zero();
  float m = -3.0e38f, lsum = 0.f;
  const float csc = 1.44269504f * 0.125f;        // log2(e) / sqrt(D)
  const int q_abs = q0 + ln;

  // Prologue: chunk 0 -> buffer 0.
  async_g2l_b128(lds_off(&kbuf[0][tid * 8]), kg + (size_t)tid * 8);
  async_g2l_b128(lds_off(&vbuf[0][vrow * 32 + vseg * 8]),
                 vg + (size_t)vrow * Sn + vseg * 8);

  for (int c = 0; c < nch; ++c) {
    const int cur = c & 1;
    if (c + 1 < nch) {                           // prefetch next chunk
      const int nxt = cur ^ 1;
      async_g2l_b128(lds_off(&kbuf[nxt][tid * 8]),
                     kg + (size_t)(c + 1) * 32 * Dn + tid * 8);
      async_g2l_b128(lds_off(&vbuf[nxt][vrow * 32 + vseg * 8]),
                     vg + (size_t)vrow * Sn + (c + 1) * 32 + vseg * 8);
      wait_async_le2();                          // chunk c's 2 ops retired (in-order)
    } else {
      wait_async_0();
    }
    __syncthreads();                             // chunk c visible block-wide

    const half_t* kb = &kbuf[cur][0];
    const half_t* vb = &vbuf[cur][0];
    const int k0 = c * 32;

    // St tiles: M = key, N = query; A = Khat rows from LDS, K-dim = 64.
    v8f st0 = v8f_zero(), st1 = v8f_zero();
    st0 = wmma_f16(load_a16(kb + ln * Dn, 0, hi), qb0, st0);
    st0 = wmma_f16(load_a16(kb + ln * Dn, 32, hi), qb1, st0);
    st1 = wmma_f16(load_a16(kb + (16 + ln) * Dn, 0, hi), qb0, st1);
    st1 = wmma_f16(load_a16(kb + (16 + ln) * Dn, 32, hi), qb1, st1);

    // Causal mask + scale in log2 domain; online softmax per query.
    float t0[8], t1[8];
    float cm = -3.0e38f;
#pragma unroll
    for (int r = 0; r < 8; ++r) {
      const int key0 = k0 + r + hi * 8;          // St C-layout: VGPR r = key row
      const int key1 = key0 + 16;
      t0[r] = (key0 <= q_abs) ? st0[r] * csc : -3.0e38f;
      t1[r] = (key1 <= q_abs) ? st1[r] * csc : -3.0e38f;
      cm = fmaxf(cm, fmaxf(t0[r], t1[r]));
    }
    cm = fmaxf(cm, __shfl_xor(cm, 16, 32));      // lane pair shares the query
    const float mn    = fmaxf(m, cm);
    const float scale = exp2f(m - mn);
    m = mn;

    float rs = 0.f;
    v16h pa;                                     // P in A-operand layout (16x32)
#pragma unroll
    for (int r = 0; r < 8; ++r) {
      const float p0 = exp2f(t0[r] - mn);
      const float p1 = exp2f(t1[r] - mn);
      rs += p0 + p1;
      pa[r]     = (half_t)p0;                    // keys k0 + hi*8 + r
      pa[8 + r] = (half_t)p1;                    // keys k0+16 + hi*8 + r
    }
    rs += __shfl_xor(rs, 16, 32);
    lsum = lsum * scale + rs;

    // Rescale O; O C-layout row r is query r + hi*8 -> broadcast from lane r+hi*8.
#pragma unroll
    for (int r = 0; r < 8; ++r) {
      const float sr = __shfl(scale, hi * 8 + r, 32);
      acc0[r] *= sr; acc1[r] *= sr; acc2[r] *= sr; acc3[r] *= sr;
    }

    // O += P(16x32) x V(32x64): B[k=key][n=d] = vbuf[d][key], K-contiguous in LDS.
    acc0 = wmma_f16(pa, load_b16(vb + (0 * 16 + ln) * 32, 0, hi), acc0);
    acc1 = wmma_f16(pa, load_b16(vb + (1 * 16 + ln) * 32, 0, hi), acc1);
    acc2 = wmma_f16(pa, load_b16(vb + (2 * 16 + ln) * 32, 0, hi), acc2);
    acc3 = wmma_f16(pa, load_b16(vb + (3 * 16 + ln) * 32, 0, hi), acc3);

    __syncthreads();                             // all readers done before next DMA
  }

  // Normalize + stage in LDS, then coalesced b128 stores (row segments of Ah).
  const float inv = 1.0f / lsum;
#pragma unroll
  for (int r = 0; r < 8; ++r) {
    const float nr = __shfl(inv, hi * 8 + r, 32);
    const int ro = (r + hi * 8) * Dn + ln;
    obuf[wave][ro + 0]  = (half_t)(acc0[r] * nr);
    obuf[wave][ro + 16] = (half_t)(acc1[r] * nr);
    obuf[wave][ro + 32] = (half_t)(acc2[r] * nr);
    obuf[wave][ro + 48] = (half_t)(acc3[r] * nr);
  }
  wait_ds0();
  half_t* abase = Ah + ((size_t)b * Sn + q0) * En + (size_t)h * Dn;
#pragma unroll
  for (int k2 = 0; k2 < 4; ++k2) {
    const int j = lane + 32 * k2;                // 128 x 16B granules
    const int r = j >> 3, d = (j & 7) * 8;
    *(v8h*)(abase + (size_t)r * En + d) = *(const v8h*)&obuf[wave][r * Dn + d];
  }
}

// ---------------------------------------------------------------- output FC
// out(BS x E) = Ah(BS x E f16) x Wfc^T + bfc.
// Block = 128 rows x 64 cols (8 waves x 16 rows on the SAME columns) so the
// Wfc K-chunk (64 cols x 32 k = 4KB) is shared: async-DMAed into double-
// buffered LDS, one b128 per thread per chunk, overlapped with WMMA.
// Cuts Wfc L2 traffic 8x vs per-wave streaming (256MB -> 32MB).
__global__ void __launch_bounds__(256)
mha_fc_kernel(const half_t* __restrict__ Ah, const half_t* __restrict__ wfc,
              const float* __restrict__ bfc, float* __restrict__ out) {
  __shared__ half_t bbuf[2][64 * 32];            // [buf][col][k] 4KB each

  const int tid  = threadIdx.x;
  const int lane = tid & 31;
  const int wave = tid >> 5;
  const int cblk = blockIdx.x & 7;               // 8 column tiles
  const int rblk = blockIdx.x >> 3;              // 64 row blocks of 128
  const int row0 = rblk * 128 + wave * 16;
  const int col0 = cblk * 64;
  const int hi   = lane >> 4, ln = lane & 15;
  const int wrow = tid >> 2, wseg = tid & 3;     // Wfc copy: 4 threads per col row

  const half_t* arow = Ah + (size_t)(row0 + ln) * En;
  const half_t* wg   = wfc + (size_t)col0 * En;  // 64 rows of Wfc, K-contiguous

  v8f acc0 = v8f_zero(), acc1 = v8f_zero(), acc2 = v8f_zero(), acc3 = v8f_zero();

  // Prologue: K-chunk 0 -> buffer 0.
  async_g2l_b128(lds_off(&bbuf[0][wrow * 32 + wseg * 8]),
                 wg + (size_t)wrow * En + wseg * 8);

  const int nch = En / 32;                       // 16 chunks
  for (int c = 0; c < nch; ++c) {
    const int cur = c & 1;
    if (c + 1 < nch) {
      async_g2l_b128(lds_off(&bbuf[cur ^ 1][wrow * 32 + wseg * 8]),
                     wg + (size_t)wrow * En + (c + 1) * 32 + wseg * 8);
      wait_async_le1();                          // chunk c's op retired (in-order)
    } else {
      wait_async_0();
    }
    __syncthreads();

    const int kk = c * 32;
    const half_t* bb = &bbuf[cur][0];
    const v16h a = load_a16(arow, kk, hi);
    acc0 = wmma_f16(a, load_b16(bb + (0 * 16 + ln) * 32, 0, hi), acc0);
    acc1 = wmma_f16(a, load_b16(bb + (1 * 16 + ln) * 32, 0, hi), acc1);
    acc2 = wmma_f16(a, load_b16(bb + (2 * 16 + ln) * 32, 0, hi), acc2);
    acc3 = wmma_f16(a, load_b16(bb + (3 * 16 + ln) * 32, 0, hi), acc3);

    __syncthreads();                             // readers done before next DMA
  }

  const float b0 = bfc[col0 + 0 * 16 + ln];
  const float b1 = bfc[col0 + 1 * 16 + ln];
  const float b2 = bfc[col0 + 2 * 16 + ln];
  const float b3 = bfc[col0 + 3 * 16 + ln];

  float* op = out + (size_t)row0 * En + col0 + ln;
#pragma unroll
  for (int r = 0; r < 8; ++r) {
    const size_t ro = (size_t)(r + hi * 8) * En;
    op[ro + 0]  = acc0[r] + b0;
    op[ro + 16] = acc1[r] + b1;
    op[ro + 32] = acc2[r] + b2;
    op[ro + 48] = acc3[r] + b3;
  }
}

// ---------------------------------------------------------------- launcher
extern "C" void kernel_launch(void* const* d_in, const int* in_sizes, int n_in,
                              void* d_out, int out_size, void* d_ws, size_t ws_size,
                              hipStream_t stream) {
  const float* qin = (const float*)d_in[0];
  const float* kin = (const float*)d_in[1];
  const float* vin = (const float*)d_in[2];
  const float* Wq  = (const float*)d_in[3];
  const float* Wk  = (const float*)d_in[4];
  const float* Wv  = (const float*)d_in[5];
  const float* Wfc = (const float*)d_in[6];
  const float* bfc = (const float*)d_in[7];
  // d_in[8] is the tril mask; causality is computed analytically in-kernel.

  char* ws = (char*)d_ws;
  half_t* wq16  = (half_t*)ws; ws += (size_t)Dn * Dn * 2;
  half_t* wk16  = (half_t*)ws; ws += (size_t)Dn * Dn * 2;
  half_t* wv16  = (half_t*)ws; ws += (size_t)Dn * Dn * 2;
  half_t* wfc16 = (half_t*)ws; ws += (size_t)En * En * 2;
  half_t* Qh    = (half_t*)ws; ws += (size_t)Bn * Hn * Sn * Dn * 2;
  half_t* Kh    = (half_t*)ws; ws += (size_t)Bn * Hn * Sn * Dn * 2;
  half_t* Vt    = (half_t*)ws; ws += (size_t)Bn * Hn * Sn * Dn * 2;
  half_t* Ah    = (half_t*)ws; ws += (size_t)Bn * Sn * En * 2;

  mha_prep_kernel<<<dim3((En * En + 255) / 256), dim3(256), 0, stream>>>(
      Wq, Wk, Wv, Wfc, wq16, wk16, wv16, wfc16);

  // Bn*Hn*(Sn/16) = 4096 wave-tiles, 8 waves per block.
  mha_proj_kernel<<<dim3(512), dim3(256), 0, stream>>>(
      qin, kin, vin, wq16, wk16, wv16, Qh, Kh, Vt);

  // Bn*Hn*(Sn/128) = 512 blocks, each owning 128 queries of one (b,h).
  mha_attn_kernel<<<dim3(512), dim3(256), 0, stream>>>(Qh, Kh, Vt, Ah);

  // (BS/128 rows) x (E/64 cols) = 64 x 8 = 512 blocks.
  mha_fc_kernel<<<dim3(512), dim3(256), 0, stream>>>(Ah, wfc16, bfc, (float*)d_out);
}
